// PNA_52183852646440
// MI455X (gfx1250) — matrix operand order repaired and checked by reading, not compiled, not run on hardware
//
#include <hip/hip_runtime.h>
#include <hip/hip_bf16.h>
#include <math.h>

// ---------------- problem constants (from reference) ----------------
#define N_NODES 50000          // 50000 = 16 * 3125  -> exact 16-row tiling
#define F_FEAT  128
#define F_OUT2  64
#define EDGES   600000
#define AVG_DEG_LOG 2.5649493574615367f   // log(13)
#define BN_EPS  1e-5f
#define STD_EPS 1e-5f

typedef __attribute__((ext_vector_type(16))) __bf16 v16bf;
typedef __attribute__((ext_vector_type(8)))  __bf16 v8bf;
typedef __attribute__((ext_vector_type(8)))  float  v8f;

// ---------------- weight packing -------------------------------------
// Pack a row-major f32 weight matrix W[K x Nout] into bf16 laid out in the
// wave32 B-fragment order this file uses (k contiguous per half) for v_wmma_f32_16x16x32_bf16:
//   out[ (kb*Nout + n)*32 + half*16 + i ]  where k = kb*32 + half*16 + i
// A lane (col n, half) then reads its whole fragment as one contiguous
// 32-byte chunk (2x global_load_b128), no per-element converts.
__global__ void pack_w(const float* __restrict__ W, __bf16* __restrict__ out,
                       int K, int Nout) {
  int idx = blockIdx.x * 256 + threadIdx.x;
  if (idx >= K * Nout) return;
  int k = idx / Nout, n = idx % Nout;
  int kb = k >> 5, r = k & 31;
  out[((size_t)kb * Nout + n) * 32 + (r >> 4) * 16 + (r & 15)] =
      (__bf16)W[(size_t)k * Nout + n];
}

// ---------------- fragment helpers -----------------------------------
// A 16x32 bf16 fragment: lanes 0-15 row M=lane,
// elems 0..15 -> K=16*half+0..15 (matches packed B order).
// With bf16 row-major activations these are two contiguous 16B loads.
__device__ __forceinline__ v16bf load_a_frag(const __bf16* __restrict__ X,
                                             int ldx, int row, int k0, int half) {
  const __bf16* xr = X + (size_t)row * ldx + k0;
  v8bf lo = *(const v8bf*)(xr + 16 * half);
  v8bf hi = *(const v8bf*)(xr + 16 * half + 8);
  return __builtin_shufflevector(lo, hi, 0, 1, 2, 3, 4, 5, 6, 7,
                                 8, 9, 10, 11, 12, 13, 14, 15);
}

// Accumulate one K-segment into a 16x16 tile (packed weights, bf16 acts).
__device__ __forceinline__ v8f wmma_seg_pk(v8f acc, const __bf16* __restrict__ X,
                                           int ldx, const __bf16* __restrict__ Wpk,
                                           int Nout, int K, int row, int col,
                                           int half) {
  const __bf16* wpl = Wpk + ((size_t)col * 2 + half) * 16;
  const size_t wstride = (size_t)Nout * 32;
  for (int kb = 0; kb < (K >> 5); ++kb) {
    v16bf b = *(const v16bf*)(wpl + (size_t)kb * wstride);
    v16bf a = load_a_frag(X, ldx, row, kb << 5, half);
    acc = __builtin_amdgcn_wmma_f32_16x16x32_bf16(false, a, false, b,
                                                  (short)0, acc, false, false);
  }
  return acc;
}

// ---------------- GEMM: Y = X @ W (+ bias), 64x16 tile per wave ------
// 4 row-subtiles share each B fragment (4x less B traffic) and give four
// independent WMMA dependency chains. Row clamping handles 3125 % 4 != 0
// (uniform per wave; duplicate subtiles simply skip their stores).
__global__ __launch_bounds__(256) void gemm16b(
    const __bf16* __restrict__ X, int ldx, const __bf16* __restrict__ Wpk,
    const float* __restrict__ bias, float* __restrict__ Y, int ldy,
    int rowGroups, int colTiles, int K, int nRows) {
  int lane = threadIdx.x;                              // 0..31
  int tile = blockIdx.x * blockDim.y + threadIdx.y;    // wave-uniform
  if (tile >= rowGroups * colTiles) return;            // uniform branch
  int rg = tile / colTiles, ct = tile % colTiles;
  int half = lane >> 4, l = lane & 15;
  int col = ct * 16 + l;
  int Nout = colTiles * 16;

  int rb[4];
#pragma unroll
  for (int r = 0; r < 4; ++r) {
    int rr = rg * 64 + r * 16;
    rb[r] = rr <= nRows - 16 ? rr : nRows - 16;        // clamp (uniform)
  }

  v8f acc0 = {}, acc1 = {}, acc2 = {}, acc3 = {};
  const __bf16* wpl = Wpk + ((size_t)col * 2 + half) * 16;
  const size_t wstride = (size_t)Nout * 32;
  for (int kb = 0; kb < (K >> 5); ++kb) {
    v16bf b = *(const v16bf*)(wpl + (size_t)kb * wstride);
    int k0 = kb << 5;
    v16bf a0 = load_a_frag(X, ldx, rb[0] + l, k0, half);
    acc0 = __builtin_amdgcn_wmma_f32_16x16x32_bf16(false, a0, false, b, (short)0, acc0, false, false);
    v16bf a1 = load_a_frag(X, ldx, rb[1] + l, k0, half);
    acc1 = __builtin_amdgcn_wmma_f32_16x16x32_bf16(false, a1, false, b, (short)0, acc1, false, false);
    v16bf a2 = load_a_frag(X, ldx, rb[2] + l, k0, half);
    acc2 = __builtin_amdgcn_wmma_f32_16x16x32_bf16(false, a2, false, b, (short)0, acc2, false, false);
    v16bf a3 = load_a_frag(X, ldx, rb[3] + l, k0, half);
    acc3 = __builtin_amdgcn_wmma_f32_16x16x32_bf16(false, a3, false, b, (short)0, acc3, false, false);
  }

  float bv = bias ? bias[col] : 0.0f;
  v8f accs[4] = {acc0, acc1, acc2, acc3};
#pragma unroll
  for (int r = 0; r < 4; ++r) {
    if (rg * 64 + r * 16 <= nRows - 16) {              // skip clamped dups
      int rbase = rb[r] + half * 8;
#pragma unroll
      for (int i = 0; i < 8; ++i)
        Y[(size_t)(rbase + i) * ldy + col] = accs[r][i] + bv;
    }
  }
}

// ---------------- fused PNA "post" GEMM ------------------------------
// Ybf = bf16( x@W0 + sum_a aggr_a@W(1..4) + amp .* (sum_a aggr_a@W(5..8))
//                                         + att .* (sum_a aggr_a@W(9..12)) + bias )
// Wpk is the packed-bf16 post weight; segment s starts at s*F*Fout elems.
// Output is bf16 because its only consumer is the following lin GEMM.
__global__ __launch_bounds__(256) void post_fused(
    const __bf16* __restrict__ Xin,  const __bf16* __restrict__ mean,
    const __bf16* __restrict__ mn,   const __bf16* __restrict__ mx,
    const __bf16* __restrict__ stdv, const float* __restrict__ amp,
    const float* __restrict__ att,   const __bf16* __restrict__ Wpk,
    const float* __restrict__ bias,  __bf16* __restrict__ Ybf,
    int F, int Fout, int rowTiles, int colTiles) {
  int lane = threadIdx.x;
  int tile = blockIdx.x * blockDim.y + threadIdx.y;
  if (tile >= rowTiles * colTiles) return;
  int rt = tile / colTiles, ct = tile % colTiles;
  int half = lane >> 4, l = lane & 15;
  int row = rt * 16 + l;
  int col = ct * 16 + l;
  size_t blk = (size_t)F * Fout;   // packed elems per K=F weight block

  v8f accA = {};                                   // x + identity-scaler aggr
  accA = wmma_seg_pk(accA, Xin,  F, Wpk + 0 * blk,  Fout, F, row, col, half);
  accA = wmma_seg_pk(accA, mean, F, Wpk + 1 * blk,  Fout, F, row, col, half);
  accA = wmma_seg_pk(accA, mn,   F, Wpk + 2 * blk,  Fout, F, row, col, half);
  accA = wmma_seg_pk(accA, mx,   F, Wpk + 3 * blk,  Fout, F, row, col, half);
  accA = wmma_seg_pk(accA, stdv, F, Wpk + 4 * blk,  Fout, F, row, col, half);
  v8f accB = {};                                   // amplification-scaled
  accB = wmma_seg_pk(accB, mean, F, Wpk + 5 * blk,  Fout, F, row, col, half);
  accB = wmma_seg_pk(accB, mn,   F, Wpk + 6 * blk,  Fout, F, row, col, half);
  accB = wmma_seg_pk(accB, mx,   F, Wpk + 7 * blk,  Fout, F, row, col, half);
  accB = wmma_seg_pk(accB, stdv, F, Wpk + 8 * blk,  Fout, F, row, col, half);
  v8f accC = {};                                   // attenuation-scaled
  accC = wmma_seg_pk(accC, mean, F, Wpk + 9 * blk,  Fout, F, row, col, half);
  accC = wmma_seg_pk(accC, mn,   F, Wpk + 10 * blk, Fout, F, row, col, half);
  accC = wmma_seg_pk(accC, mx,   F, Wpk + 11 * blk, Fout, F, row, col, half);
  accC = wmma_seg_pk(accC, stdv, F, Wpk + 12 * blk, Fout, F, row, col, half);

  float bv = bias[col];
  int rbase = rt * 16 + half * 8;
#pragma unroll
  for (int i = 0; i < 8; ++i) {
    int r = rbase + i;
    Ybf[(size_t)r * Fout + col] =
        (__bf16)(accA[i] + amp[r] * accB[i] + att[r] * accC[i] + bv);
  }
}

// ---------------- monotonic float<->uint encoding for atomic min/max -
__device__ __forceinline__ unsigned fenc(float x) {
  unsigned u = __float_as_uint(x);
  return (u & 0x80000000u) ? ~u : (u | 0x80000000u);
}
__device__ __forceinline__ float fdec(unsigned e) {
  unsigned u = (e & 0x80000000u) ? (e & 0x7FFFFFFFu) : ~e;
  return __uint_as_float(u);
}

// ---------------- small utility kernels ------------------------------
__global__ void fill_u32(unsigned* p, unsigned v, size_t n) {
  size_t i = (size_t)blockIdx.x * blockDim.x + threadIdx.x;
  if (i < n) p[i] = v;
}

__global__ void f32_to_bf16(const float* __restrict__ in, __bf16* __restrict__ out) {
  size_t i = (size_t)blockIdx.x * 256 + threadIdx.x;   // exact grid
  out[i] = (__bf16)in[i];
}

__global__ void degree_kernel(const long long* __restrict__ ei, int* __restrict__ cnt) {
  int e = blockIdx.x * blockDim.x + threadIdx.x;
  if (e >= EDGES) return;
  int d = (int)ei[(size_t)EDGES + e];     // dst = edge_index[1]
  atomicAdd(&cnt[d], 1);
}

__global__ void scalers_kernel(const int* __restrict__ cnt,
                               float* __restrict__ amp, float* __restrict__ att) {
  int i = blockIdx.x * blockDim.x + threadIdx.x;
  if (i >= N_NODES) return;
  float deg = fmaxf((float)cnt[i], 1.0f);
  float l = logf(deg + 1.0f);
  amp[i] = l / AVG_DEG_LOG;
  att[i] = AVG_DEG_LOG / l;
}

// Edge pass, 4 features per thread: h = A[dst] + B[src] (A,B are 25.6MB
// each -> L2-resident gathers as b128), then f32 add atomics and
// native u32 min/max atomics with the monotonic encoding.
__global__ __launch_bounds__(256) void edge_agg4(
    const long long* __restrict__ ei,
    const float* __restrict__ A, const float* __restrict__ B,
    float* __restrict__ sum, float* __restrict__ sumsq,
    unsigned* __restrict__ mnE, unsigned* __restrict__ mxE) {
  unsigned idx = blockIdx.x * 256u + threadIdx.x;   // exact: EDGES*32 threads
  unsigned e = idx >> 5;
  unsigned q = (idx & 31u) << 2;                    // feature quad base
  int s = (int)ei[e];
  int d = (int)ei[(size_t)EDGES + e];
  const float4 av = *(const float4*)&A[((size_t)d << 7) + q];
  const float4 bv = *(const float4*)&B[((size_t)s << 7) + q];
  float h[4] = {av.x + bv.x, av.y + bv.y, av.z + bv.z, av.w + bv.w};
  size_t o = ((size_t)d << 7) + q;
#pragma unroll
  for (int j = 0; j < 4; ++j) {
    atomicAdd(&sum[o + j], h[j]);
    atomicAdd(&sumsq[o + j], h[j] * h[j]);
    unsigned enc = fenc(h[j]);
    atomicMax(&mxE[o + j], enc);
    atomicMin(&mnE[o + j], enc);
  }
}

// mean/std/min/max finalized straight into bf16 (the WMMA operand format);
// the f32 atomic buffers are just scratch after this point.
__global__ __launch_bounds__(256) void finalize_kernel(
    const float* __restrict__ sum, const float* __restrict__ sumsq,
    const unsigned* __restrict__ mnE, const unsigned* __restrict__ mxE,
    const int* __restrict__ cnt,
    __bf16* __restrict__ meanb, __bf16* __restrict__ stdb,
    __bf16* __restrict__ mnb, __bf16* __restrict__ mxb) {
  size_t idx = (size_t)blockIdx.x * 256 + threadIdx.x;   // exact: N*F
  int node = (int)(idx >> 7);
  int c = cnt[node];
  float deg = fmaxf((float)c, 1.0f);
  float m = sum[idx] / deg;
  float ms = sumsq[idx] / deg;
  float var = fmaxf(ms - m * m, 0.0f);
  bool has = c > 0;
  meanb[idx] = (__bf16)m;
  stdb[idx]  = (__bf16)sqrtf(var + STD_EPS);
  mnb[idx]   = (__bf16)(has ? fdec(mnE[idx]) : 0.0f);
  mxb[idx]   = (__bf16)(has ? fdec(mxE[idx]) : 0.0f);
}

// BN(eval) + ReLU, emitted directly as bf16 (layer-2 WMMA operand).
__global__ __launch_bounds__(256) void bn_relu_kernel(
    const float* __restrict__ h, __bf16* __restrict__ outb,
    const float* __restrict__ g, const float* __restrict__ b,
    const float* __restrict__ m, const float* __restrict__ v) {
  size_t idx = (size_t)blockIdx.x * 256 + threadIdx.x;   // exact: N*F
  int f = (int)(idx & 127u);
  float y = (h[idx] - m[f]) * rsqrtf(v[f] + BN_EPS) * g[f] + b[f];
  outb[idx] = (__bf16)fmaxf(y, 0.0f);
}

// ---------------- host orchestration ---------------------------------
extern "C" void kernel_launch(void* const* d_in, const int* in_sizes, int n_in,
                              void* d_out, int out_size, void* d_ws, size_t ws_size,
                              hipStream_t stream) {
  (void)in_sizes; (void)n_in; (void)out_size; (void)ws_size;
  const float*      x      = (const float*)d_in[0];
  const long long*  ei     = (const long long*)d_in[1];   // int64 [2,E]
  const float* preW1  = (const float*)d_in[2];
  const float* preb1  = (const float*)d_in[3];
  const float* postW1 = (const float*)d_in[4];
  const float* postb1 = (const float*)d_in[5];
  const float* linW1  = (const float*)d_in[6];
  const float* linb1  = (const float*)d_in[7];
  const float* bng    = (const float*)d_in[8];
  const float* bnb    = (const float*)d_in[9];
  const float* bnm    = (const float*)d_in[10];
  const float* bnv    = (const float*)d_in[11];
  const float* preW2  = (const float*)d_in[12];
  const float* preb2  = (const float*)d_in[13];
  const float* postW2 = (const float*)d_in[14];
  const float* postb2 = (const float*)d_in[15];
  const float* linW2  = (const float*)d_in[16];
  const float* linb2  = (const float*)d_in[17];
  float* out = (float*)d_out;

  // workspace carve-out (buffers reused across the two layers): ~232 MB
  char* wptr = (char*)d_ws;
  auto take = [&](size_t bytes) -> void* {
    void* p = (void*)wptr; wptr += (bytes + 255) & ~(size_t)255; return p;
  };
  const size_t NF = (size_t)N_NODES * F_FEAT;
  float*    A     = (float*)take(NF * 4);   // pre-out A; later lin1 f32 out
  float*    Bm    = (float*)take(NF * 4);
  float*    sum   = (float*)take(NF * 4);
  float*    sumsq = (float*)take(NF * 4);
  unsigned* mnE   = (unsigned*)take(NF * 4);
  unsigned* mxE   = (unsigned*)take(NF * 4);
  // bf16 WMMA activation operands
  __bf16* xbf    = (__bf16*)take(NF * 2);   // layer input (x, then relu(h1))
  __bf16* meanb  = (__bf16*)take(NF * 2);
  __bf16* stdb   = (__bf16*)take(NF * 2);
  __bf16* mnb    = (__bf16*)take(NF * 2);
  __bf16* mxb    = (__bf16*)take(NF * 2);
  __bf16* postb_ = (__bf16*)take(NF * 2);
  int*    cnt    = (int*)take((size_t)N_NODES * 4);
  float*  amp    = (float*)take((size_t)N_NODES * 4);
  float*  att    = (float*)take((size_t)N_NODES * 4);
  // packed bf16 weights (~0.8 MB total)
  __bf16* preP1  = (__bf16*)take((size_t)256 * 128 * 2);
  __bf16* postP1 = (__bf16*)take((size_t)1664 * 128 * 2);
  __bf16* linP1  = (__bf16*)take((size_t)128 * 128 * 2);
  __bf16* preP2  = (__bf16*)take((size_t)256 * 128 * 2);
  __bf16* postP2 = (__bf16*)take((size_t)1664 * 64 * 2);
  __bf16* linP2  = (__bf16*)take((size_t)64 * 64 * 2);

  const dim3 gblk(32, 8);                       // 8 waves / block
  const int rowTiles = N_NODES / 16;            // 3125, exact
  const int rowGroups = (rowTiles + 3) / 4;     // 782 (64-row groups)
  const int fillBlkNF = (int)((NF + 255) / 256);

  // pack all weights into WMMA B-fragment order (re-done every call)
  auto pack = [&](const float* W, __bf16* P, int K, int Nout) {
    pack_w<<<(K * Nout + 255) / 256, 256, 0, stream>>>(W, P, K, Nout);
  };
  pack(preW1, preP1, 256, 128);   pack(postW1, postP1, 1664, 128);
  pack(linW1, linP1, 128, 128);   pack(preW2, preP2, 256, 128);
  pack(postW2, postP2, 1664, 64); pack(linW2, linP2, 64, 64);

  // bf16 copy of the input features
  f32_to_bf16<<<(int)(NF / 256), 256, 0, stream>>>(x, xbf);

  // degree + scalers (edge structure is layer-invariant)
  fill_u32<<<(N_NODES + 255) / 256, 256, 0, stream>>>((unsigned*)cnt, 0u, N_NODES);
  degree_kernel<<<(EDGES + 255) / 256, 256, 0, stream>>>(ei, cnt);
  scalers_kernel<<<(N_NODES + 255) / 256, 256, 0, stream>>>(cnt, amp, att);

  auto gemm = [&](const __bf16* X, int ldx, const __bf16* Wpk, const float* bias,
                  float* Y, int Nout, int K) {
    int ct = Nout / 16, tiles = rowGroups * ct;
    gemm16b<<<(tiles + 7) / 8, gblk, 0, stream>>>(X, ldx, Wpk, bias, Y, Nout,
                                                  rowGroups, ct, K, N_NODES);
  };

  auto run_layer = [&](const __bf16* Xin, const __bf16* preP, const float* preb,
                       const __bf16* postP, const float* postb,
                       const __bf16* linP, const float* linb,
                       int Fout, float* linOut) {
    // pre-MLP hoisted to node level: A = Xin@W_top + b,  B = Xin@W_bot
    // (packed preP holds both halves; bottom half starts at kb=4 -> 128*128)
    gemm(Xin, F_FEAT, preP, preb, A, F_FEAT, F_FEAT);
    gemm(Xin, F_FEAT, preP + (size_t)128 * 128, nullptr, Bm, F_FEAT, F_FEAT);
    // aggregation buffers
    fill_u32<<<fillBlkNF, 256, 0, stream>>>((unsigned*)sum, 0u, NF);
    fill_u32<<<fillBlkNF, 256, 0, stream>>>((unsigned*)sumsq, 0u, NF);
    fill_u32<<<fillBlkNF, 256, 0, stream>>>(mnE, 0xFFFFFFFFu, NF);
    fill_u32<<<fillBlkNF, 256, 0, stream>>>(mxE, 0u, NF);
    // edge pass (exact grid: E*32 / 256 = 75000 blocks)
    edge_agg4<<<(EDGES * 32) / 256, 256, 0, stream>>>(ei, A, Bm, sum, sumsq, mnE, mxE);
    finalize_kernel<<<(int)(NF / 256), 256, 0, stream>>>(sum, sumsq, mnE, mxE, cnt,
                                                         meanb, stdb, mnb, mxb);
    // fused post-MLP with scaler epilogue -> bf16
    {
      int ct = Fout / 16, tiles = rowTiles * ct;
      post_fused<<<(tiles + 7) / 8, gblk, 0, stream>>>(
          Xin, meanb, mnb, mxb, stdb, amp, att, postP, postb, postb_,
          F_FEAT, Fout, rowTiles, ct);
    }
    // final linear (f32 out)
    gemm(postb_, Fout, linP, linb, linOut, Fout, Fout);
  };

  // layer 1: lin output goes into A (free at that point), then BN+ReLU -> xbf
  run_layer(xbf, preP1, preb1, postP1, postb1, linP1, linb1, F_FEAT, A);
  bn_relu_kernel<<<(int)(NF / 256), 256, 0, stream>>>(A, xbf, bng, bnb, bnm, bnv);
  // layer 2: final result straight to d_out
  run_layer(xbf, preP2, preb2, postP2, postb2, linP2, linb2, F_OUT2, out);
}
